// PolicyHead_11862699672098
// MI455X (gfx1250) — compile-verified, matrix operand order checked by reading, not run on hardware
//
#include <hip/hip_runtime.h>

typedef __attribute__((ext_vector_type(16))) __bf16 v16bf;
typedef __attribute__((ext_vector_type(8)))  __bf16 v8bf;
typedef __attribute__((ext_vector_type(4)))  __bf16 v4bf;
typedef __attribute__((ext_vector_type(8)))  float  v8f;

#define H 256
#define NTOK 64
#define SCALE 0.0625f
#define N_POLICY 1858

// ---------------------------------------------------------------------------
// Load a 16x32 bf16 A/B fragment (ISA 7.12.2 16-bit layout) from a row-major
// [rows x 256] bf16 array. Lane l: half = l>>4, row = base + (l&15).
// Elems 0..7:  k = kbase + half*8 .. +7
// Elems 8..15: k = kbase + 16 + half*8 .. +7      -> two b128 loads.
// ---------------------------------------------------------------------------
__device__ __forceinline__ v16bf load_frag(const __bf16* __restrict__ base,
                                           int row, int kbase, int half) {
    const __bf16* p = base + row * H + kbase + (half << 3);
    v8bf lo = *(const v8bf*)p;
    v8bf hi = *(const v8bf*)(p + 16);
    v16bf f;
#pragma unroll
    for (int i = 0; i < 8; ++i) { f[i] = lo[i]; f[8 + i] = hi[i]; }
    return f;
}

// ---------------------------------------------------------------------------
// 64x256 = (64x256) @ (256x256)^T GEMM. A in LDS (bf16), W global bf16
// row-major [n][k], bias fp32, optional relu, result -> LDS bf16 [64][256].
// Each wave owns 2 N-column tiles x ALL 4 M-tiles: every global B-fragment
// is loaded once and reused by 4 WMMAs (A-fragments streamed from LDS).
// Weight matrix is read exactly once per workgroup.
// ---------------------------------------------------------------------------
__device__ __forceinline__ void gemm_64x256(const __bf16* __restrict__ A,
                                            const __bf16* __restrict__ W,
                                            const float* __restrict__ bias,
                                            __bf16* __restrict__ Out,
                                            bool relu, int tid) {
    const int lane = tid & 31;
    const int wave = tid >> 5;
    const int half = lane >> 4;
    const int lrow = lane & 15;
#pragma unroll
    for (int c = 0; c < 2; ++c) {
        const int nT = wave + c * 8;          // 0..15
        const int n  = nT * 16 + lrow;
        v8f acc[4] = {v8f{}, v8f{}, v8f{}, v8f{}};
#pragma unroll
        for (int ks = 0; ks < 8; ++ks) {
            const v16bf b = load_frag(W, n, ks * 32, half);   // global, once
#pragma unroll
            for (int mT = 0; mT < 4; ++mT) {
                v16bf a = load_frag(A, mT * 16 + lrow, ks * 32, half); // LDS
                acc[mT] = __builtin_amdgcn_wmma_f32_16x16x32_bf16(
                    false, a, false, b, (short)0, acc[mT], false, false);
            }
        }
        const float bs = bias[n];
#pragma unroll
        for (int mT = 0; mT < 4; ++mT) {
#pragma unroll
            for (int r = 0; r < 8; ++r) {
                float v = acc[mT][r] + bs;
                if (relu) v = v > 0.f ? v : 0.f;
                const int m = mT * 16 + r + (half << 3);
                Out[m * H + n] = (__bf16)v;
            }
        }
    }
}

// ---------------------------------------------------------------------------
// fp32 -> bf16 weight conversion (w1,w2,w3 concatenated into workspace)
// ---------------------------------------------------------------------------
__global__ void convert_weights(const float* __restrict__ w1,
                                const float* __restrict__ w2,
                                const float* __restrict__ w3,
                                __bf16* __restrict__ out) {
    const int i = blockIdx.x * blockDim.x + threadIdx.x;  // 0 .. 3*65536-1
    float v;
    if (i < 65536)       v = w1[i];
    else if (i < 131072) v = w2[i - 65536];
    else                 v = w3[i - 131072];
    out[i] = (__bf16)v;
}

// ---------------------------------------------------------------------------
// Fused policy head: one workgroup (256 thr = 8 waves) per board (64 tokens).
// ---------------------------------------------------------------------------
__global__ void __launch_bounds__(256, 2)
policy_head_kernel(const float* __restrict__ x,
                   const __bf16* __restrict__ w1b,
                   const __bf16* __restrict__ w2b,
                   const __bf16* __restrict__ w3b,
                   const float* __restrict__ b1,
                   const float* __restrict__ b2,
                   const float* __restrict__ b3,
                   const float* __restrict__ w4,
                   const int* __restrict__ gidx,
                   float* __restrict__ out) {
    __shared__ __align__(16) __bf16 s_out[NTOK * H];  // relu activations (32KB)
    __shared__ __align__(16) __bf16 s_k[NTOK * H];    // k               (32KB)
    __shared__ __align__(16) __bf16 s_xq[NTOK * H];   // x, then q       (32KB)
    __shared__ float s_lg[67 * 64];                   // logits+promo  (17.2KB)
    __shared__ float s_offs[32];
    __shared__ float s_off24[24];

    const int tid = threadIdx.x;
    const int b   = blockIdx.x;
    const int lane = tid & 31;
    const int wave = tid >> 5;
    const int half = lane >> 4;
    const int lrow = lane & 15;

    // ---- Stage 0: load x tile (64x256 fp32) -> bf16 LDS, float4 vectorized
    {
        const float4* xr = (const float4*)(x + (size_t)b * NTOK * H);
        for (int i = tid; i < NTOK * H / 4; i += 256) {
            float4 v = xr[i];
            const int r = i >> 6;          // 64 float4 per row
            const int c = (i & 63) << 2;
            v4bf p; p[0] = (__bf16)v.x; p[1] = (__bf16)v.y;
                    p[2] = (__bf16)v.z; p[3] = (__bf16)v.w;
            *(v4bf*)&s_xq[r * H + c] = p;
        }
    }
    __syncthreads();

    // ---- Stage 1: out = relu(x @ w1^T + b1)
    gemm_64x256(s_xq, w1b, b1, s_out, true, tid);
    __syncthreads();

    // ---- Stage 2: k = out @ w3^T + b3
    gemm_64x256(s_out, w3b, b3, s_k, false, tid);
    __syncthreads();

    // ---- Stage 3: q = out @ w2^T + b2  (reuse x buffer)
    gemm_64x256(s_out, w2b, b2, s_xq, false, tid);
    __syncthreads();

    // ---- Stage 4: logits = q @ k^T * SCALE   (64x64, K=256; 4x4 tiles)
    // wave w: mT = w>>1, nT in { (w&1)*2, (w&1)*2+1 }; A-fragment shared.
    {
        const int mT  = wave >> 1;
        const int nT0 = (wave & 1) * 2;
        v8f acc[2] = {v8f{}, v8f{}};
#pragma unroll
        for (int ks = 0; ks < 8; ++ks) {
            v16bf a = load_frag(s_xq, mT * 16 + lrow, ks * 32, half);
#pragma unroll
            for (int j = 0; j < 2; ++j) {
                v16bf bb = load_frag(s_k, (nT0 + j) * 16 + lrow, ks * 32, half);
                acc[j] = __builtin_amdgcn_wmma_f32_16x16x32_bf16(
                    false, a, false, bb, (short)0, acc[j], false, false);
            }
        }
#pragma unroll
        for (int j = 0; j < 2; ++j) {
            const int n = (nT0 + j) * 16 + lrow;
#pragma unroll
            for (int r = 0; r < 8; ++r) {
                const int m = mT * 16 + r + (half << 3);
                s_lg[m * 64 + n] = acc[j][r] * SCALE;
            }
        }
    }
    __syncthreads();

    // ---- Stage 5: promo offsets  offs[p][s] = k[56+s] . w4[p]
    if (tid < 32) {
        const int p = tid >> 3, s = tid & 7;
        const __bf16* kr = s_k + (56 + s) * H;
        const float* wr = w4 + p * H;
        float acc = 0.f;
        for (int h = 0; h < H; ++h) acc += (float)kr[h] * wr[h];
        s_offs[p * 8 + s] = acc;
    }
    __syncthreads();
    if (tid < 24) {
        const int s = tid / 3, p = tid % 3;
        s_off24[tid] = s_offs[p * 8 + s] + s_offs[3 * 8 + s];
    }
    __syncthreads();
    // promo rows 64..66:  full[64+r][c] = off24[(r*64+c)%24] + L[48+u/8][56+u%8]
    if (tid < 192) {
        const int r = tid >> 6, c = tid & 63;
        const int flat = r * 64 + c;
        const int u = flat / 3;
        s_lg[(64 + r) * 64 + c] =
            s_off24[flat % 24] + s_lg[(48 + (u >> 3)) * 64 + 56 + (u & 7)];
    }
    __syncthreads();

    // ---- Stage 6: gather 1858 policy outputs
    float* ob = out + (size_t)b * N_POLICY;
    for (int g = tid; g < N_POLICY; g += 256) {
        ob[g] = s_lg[gidx[g]];
    }
}

extern "C" void kernel_launch(void* const* d_in, const int* in_sizes, int n_in,
                              void* d_out, int out_size, void* d_ws, size_t ws_size,
                              hipStream_t stream) {
    const float* x  = (const float*)d_in[0];
    const float* w1 = (const float*)d_in[1];
    const float* b1 = (const float*)d_in[2];
    const float* w2 = (const float*)d_in[3];
    const float* b2 = (const float*)d_in[4];
    const float* w3 = (const float*)d_in[5];
    const float* b3 = (const float*)d_in[6];
    const float* w4 = (const float*)d_in[7];
    const int* gidx = (const int*)d_in[8];

    __bf16* wb = (__bf16*)d_ws;   // 3 * 65536 bf16 = 384 KB

    convert_weights<<<3 * 65536 / 256, 256, 0, stream>>>(w1, w2, w3, wb);
    policy_head_kernel<<<4096, 256, 0, stream>>>(
        x, wb, wb + 65536, wb + 2 * 65536,
        b1, b2, b3, w4, gidx, (float*)d_out);
}
